// DynamicMultiHeadConv_32117765439534
// MI455X (gfx1250) — compile-verified, matrix operand order checked by reading, not run on hardware
//
#include <hip/hip_runtime.h>
#include <cstddef>

// ---------------------------------------------------------------------------
// DynamicMultiHeadConv for MI455X (gfx1250, wave32, WMMA)
// B=16, C=256, H=W=56, HEADS=4, CPER=64, R=16, K=3, inactive=192
// ---------------------------------------------------------------------------

typedef __attribute__((ext_vector_type(16))) __bf16 v16bf;
typedef __attribute__((ext_vector_type(8)))  float  v8f;

#define NB    16
#define NC    256
#define NHW   3136      // 56*56
#define NH    56
#define HEADS 4
#define CPER  64
#define NR    16

static __device__ __forceinline__ unsigned short f2bf(float f) {
    unsigned u = __float_as_uint(f);
    unsigned r = (u + 0x7FFFu + ((u >> 16) & 1u)) >> 16;   // RNE
    return (unsigned short)r;
}

// ---------------------------------------------------------------------------
// K1: BN(+ReLU) global average pool.  grid = B*C blocks, 256 thr.
// Also stashes per-channel BN scale/bias once.
// ---------------------------------------------------------------------------
__global__ __launch_bounds__(256) void k1_gap(
    const float* __restrict__ x, const float* __restrict__ gamma,
    const float* __restrict__ beta, const float* __restrict__ mean,
    const float* __restrict__ var,
    float* __restrict__ xavg, float* __restrict__ sbn, float* __restrict__ bbn)
{
    int bc = blockIdx.x;              // b*256 + c
    int c  = bc & 255;
    int b  = bc >> 8;
    float s    = gamma[c] * rsqrtf(var[c] + 1e-5f);
    float bias = beta[c] - mean[c] * s;
    if (b == 0 && threadIdx.x == 0) { sbn[c] = s; bbn[c] = bias; }

    const float4* xp = (const float4*)(x + (size_t)bc * NHW);
    float sum = 0.f;
    for (int i = threadIdx.x; i < NHW / 4; i += 256) {
        float4 v = xp[i];
        sum += fmaxf(0.f, v.x * s + bias) + fmaxf(0.f, v.y * s + bias)
             + fmaxf(0.f, v.z * s + bias) + fmaxf(0.f, v.w * s + bias);
    }
    __shared__ float red[256];
    red[threadIdx.x] = sum;
    __syncthreads();
    for (int st = 128; st > 0; st >>= 1) {
        if (threadIdx.x < st) red[threadIdx.x] += red[threadIdx.x + st];
        __syncthreads();
    }
    if (threadIdx.x == 0) xavg[bc] = red[0] * (1.0f / (float)NHW);
}

// ---------------------------------------------------------------------------
// K2: per-(head,b) SE gate, rank-select threshold, deterministic compaction
// of active channels with pre-folded BN*mask scales.  grid = 64, 256 thr.
// ---------------------------------------------------------------------------
__global__ __launch_bounds__(256) void k2_gate(
    const float* __restrict__ xavg, const float* __restrict__ fc1,
    const float* __restrict__ fc2,  const float* __restrict__ fc2b,
    const float* __restrict__ sbn,  const float* __restrict__ bbn,
    const int* __restrict__ inact,
    float* __restrict__ lasso_part, int* __restrict__ nact_arr,
    int* __restrict__ act_idx, float* __restrict__ act_a, float* __restrict__ act_b)
{
    int lid = blockIdx.x;             // h*16 + b
    int h = lid >> 4, b = lid & 15;
    int tid = threadIdx.x;

    __shared__ float xa[NC], y1[NR], msk[NC], red[NC];
    __shared__ int   flag[NC];
    __shared__ float thr;
    __shared__ int   sh_n;

    xa[tid] = xavg[b * NC + tid];
    if (tid == 0) thr = -1e30f;
    __syncthreads();

    if (tid < NR) {
        const float* w = fc1 + ((size_t)h * NR + tid) * NC;
        float s = 0.f;
        for (int c = 0; c < NC; ++c) s += xa[c] * w[c];
        y1[tid] = fmaxf(0.f, s);
    }
    __syncthreads();

    {
        const float* w = fc2 + ((size_t)h * NC + tid) * NR;
        float s = fc2b[h * NC + tid];
        for (int r = 0; r < NR; ++r) s += y1[r] * w[r];
        msk[tid] = fmaxf(0.f, s);
    }
    red[tid] = msk[tid];
    __syncthreads();
    for (int st = 128; st > 0; st >>= 1) {
        if (tid < st) red[tid] += red[tid + st];
        __syncthreads();
    }
    if (tid == 0) lasso_part[lid] = red[0];

    // rank-(k-1) value == k-th smallest == threshold
    int k = inact[0];
    if (k > 0) {
        float v = msk[tid];
        int cl = 0, ce = 0;
        for (int j = 0; j < NC; ++j) {
            float u = msk[j];
            cl += (u < v);
            ce += (u == v);
        }
        if (cl <= k - 1 && k - 1 < cl + ce) thr = v;   // identical value if several writers
    }
    __syncthreads();
    float t = (k > 0) ? thr : -1e30f;

    int f = (msk[tid] > t) ? 1 : 0;
    flag[tid] = f;
    __syncthreads();

    int pos = 0;
    for (int j = 0; j < tid; ++j) pos += flag[j];
    if (tid == 255) sh_n = pos + f;
    __syncthreads();
    int n = sh_n;

    // disjoint slots: actives -> [0,n), defaults -> [n,256)
    if (tid >= n) {
        act_idx[lid * NC + tid] = 0;
        act_a  [lid * NC + tid] = 0.f;
        act_b  [lid * NC + tid] = 0.f;
    }
    if (f) {
        float m = msk[tid];
        act_idx[lid * NC + pos] = tid;
        act_a  [lid * NC + pos] = sbn[tid] * m;
        act_b  [lid * NC + pos] = bbn[tid] * m;
    }
    if (tid == 0) nact_arr[lid] = n;
}

// ---------------------------------------------------------------------------
// K3: lasso loss, fixed summation order.
// ---------------------------------------------------------------------------
__global__ void k3_lasso(const float* __restrict__ part, float* __restrict__ outs)
{
    if (threadIdx.x == 0 && blockIdx.x == 0) {
        float s = 0.f;
        for (int i = 0; i < HEADS * NB; ++i) s += part[i];
        outs[0] = s * (1.0f / (float)(NB * NC));   // sum_h mean_{b,c}
    }
}

// ---------------------------------------------------------------------------
// K4: sparse implicit-GEMM grouped conv via bf16 WMMA, double-buffered LDS.
// grid = B*HEADS*16 tiles, 256 thr (8 waves).
// Block: one (b, head, 16x16 spatial tile).  M=256 positions, N=64 outch,
// K = 32 active channels per chunk x 9 taps.  Stage chunk ck+1 while
// computing chunk ck (one barrier per chunk).
// ---------------------------------------------------------------------------
__global__ __launch_bounds__(256) void k4_conv(
    const float* __restrict__ x, const float* __restrict__ conv_w,
    const int* __restrict__ nact_arr, const int* __restrict__ act_idx,
    const float* __restrict__ act_a, const float* __restrict__ act_b,
    float* __restrict__ out)
{
    // double-buffered: 18x18 patch x 32 ch bf16 (pairs in u32), and
    // (tap*64+oc) x 32 ch bf16 weights.  2*(20736+36864) = 115200 B LDS.
    __shared__ __align__(16) unsigned int A_lds[2][324 * 16];
    __shared__ __align__(16) unsigned int B_lds[2][9 * 64 * 16];
    __shared__ int   s_gc[2][32];
    __shared__ float s_a[2][32], s_b[2][32];

    int blk  = blockIdx.x;
    int tile = blk & 15;
    int lid  = blk >> 4;             // h*16 + b  (matches K2)
    int h = lid >> 4, b = lid & 15;
    int ty0 = (tile >> 2) * 16, tx0 = (tile & 3) * 16;

    int tid  = threadIdx.x;
    int lane = tid & 31;
    int wave = tid >> 5;
    int n15  = lane & 15;
    int klo2 = (lane & 16) ? 4 : 0;  // u32 offset of K-low half per documented layout

    union Frag { uint4 q[2]; v16bf v; };

    v8f zero = {};
    v8f acc[2][4];
#pragma unroll
    for (int mt = 0; mt < 2; ++mt)
#pragma unroll
        for (int nt = 0; nt < 4; ++nt) acc[mt][nt] = zero;

    int nact = nact_arr[lid];
    int nch  = (nact + 31) >> 5;

    const float* xb_base = x + (size_t)b * (NC * NHW);
    const float* wh_base = conv_w + (size_t)h * (CPER * NC * 9);
    int py0 = (ty0 > 0) ? (ty0 - 1) : 0;

    // --- helpers -----------------------------------------------------------
    auto load_meta = [&](int chunk, int slot) {
        int s0 = chunk * 32 + tid;
        int gc = act_idx[lid * NC + s0];
        s_gc[slot][tid] = gc;
        s_a[slot][tid]  = act_a[lid * NC + s0];
        s_b[slot][tid]  = act_b[lid * NC + s0];
        // warm L2 for the upcoming gathered patch of this channel plane
        __builtin_prefetch(xb_base + (long)gc * NHW + (long)py0 * NH + tx0, 0, 3);
    };

    auto stage = [&](int buf) {
        // ---- activations: BN+ReLU+mask fused, fp32 -> bf16 ----
        for (int p = tid; p < 324; p += 256) {
            int py = p / 18, px = p - py * 18;
            int gy = ty0 + py - 1, gx = tx0 + px - 1;
            bool inb = (gy >= 0) && (gy < NH) && (gx >= 0) && (gx < NH);
            long off = (long)gy * NH + gx;
#pragma unroll
            for (int jj = 0; jj < 32; jj += 2) {
                float v0 = 0.f, v1 = 0.f;
                if (inb) {
                    float r0 = xb_base[(long)s_gc[buf][jj]     * NHW + off];
                    float r1 = xb_base[(long)s_gc[buf][jj + 1] * NHW + off];
                    v0 = fmaxf(0.f, r0 * s_a[buf][jj]     + s_b[buf][jj]);
                    v1 = fmaxf(0.f, r1 * s_a[buf][jj + 1] + s_b[buf][jj + 1]);
                }
                A_lds[buf][p * 16 + (jj >> 1)] =
                    (unsigned)f2bf(v0) | ((unsigned)f2bf(v1) << 16);
            }
        }
        // ---- gathered weights: fp32 -> bf16 ----
        for (int e = tid; e < 9216; e += 256) {
            int chp  = e & 15;
            int rest = e >> 4;             // tap*64 + oc
            int oc   = rest & 63, tap = rest >> 6;
            int c0 = s_gc[buf][chp * 2], c1 = s_gc[buf][chp * 2 + 1];
            const float* wr = wh_base + (size_t)oc * (NC * 9) + tap;
            float w0 = wr[c0 * 9], w1 = wr[c1 * 9];
            B_lds[buf][e] = (unsigned)f2bf(w0) | ((unsigned)f2bf(w1) << 16);
        }
    };

    auto compute = [&](int buf) {
#pragma unroll
        for (int tap = 0; tap < 9; ++tap) {
            int ky = tap / 3, kx = tap - ky * 3;
            Frag bfr[4];
#pragma unroll
            for (int nt = 0; nt < 4; ++nt) {
                int row = (tap * 64 + nt * 16 + n15) * 16;
                bfr[nt].q[0] = *(const uint4*)&B_lds[buf][row + klo2];
                bfr[nt].q[1] = *(const uint4*)&B_lds[buf][row + klo2 + 8];
            }
            Frag afr[2];
#pragma unroll
            for (int mt = 0; mt < 2; ++mt) {
                int my = wave * 2 + mt;
                int p  = (my + ky) * 18 + n15 + kx;
                afr[mt].q[0] = *(const uint4*)&A_lds[buf][p * 16 + klo2];
                afr[mt].q[1] = *(const uint4*)&A_lds[buf][p * 16 + klo2 + 8];
            }
#pragma unroll
            for (int mt = 0; mt < 2; ++mt)
#pragma unroll
                for (int nt = 0; nt < 4; ++nt)
                    acc[mt][nt] = __builtin_amdgcn_wmma_f32_16x16x32_bf16(
                        false, afr[mt].v, false, bfr[nt].v,
                        (short)0, acc[mt][nt], false, false);
        }
    };

    // --- software pipeline -------------------------------------------------
    if (nch > 0) {
        if (tid < 32) load_meta(0, 0);
        __syncthreads();
        stage(0);
    }
    for (int ck = 0; ck < nch; ++ck) {
        int cur  = ck & 1;
        int nxt  = cur ^ 1;
        bool more = (ck + 1) < nch;
        if (more && tid < 32) load_meta(ck + 1, nxt);
        __syncthreads();            // buf[cur] staged; meta[nxt] visible
        if (more) stage(nxt);       // overlaps with WMMAs below
        compute(cur);
    }

    // ---- store, channel-interleaved: final channel = oc*HEADS + h ----
    int mhi = (lane >> 4) << 3;               // 0 or 8
    float* outb = out + (size_t)b * (NC * NHW);
#pragma unroll
    for (int mt = 0; mt < 2; ++mt) {
        int gy = ty0 + wave * 2 + mt;
        if (gy >= NH) continue;
#pragma unroll
        for (int nt = 0; nt < 4; ++nt) {
            int cf = (nt * 16 + n15) * HEADS + h;
            float* op = outb + (size_t)cf * NHW + gy * NH;
#pragma unroll
            for (int r = 0; r < 8; ++r) {
                int gx = tx0 + mhi + r;
                if (gx < NH) op[gx] = acc[mt][nt][r];
            }
        }
    }
}

// ---------------------------------------------------------------------------
extern "C" void kernel_launch(void* const* d_in, const int* in_sizes, int n_in,
                              void* d_out, int out_size, void* d_ws, size_t ws_size,
                              hipStream_t stream)
{
    const float* x        = (const float*)d_in[0];
    const float* bn_gamma = (const float*)d_in[1];
    const float* bn_beta  = (const float*)d_in[2];
    const float* bn_mean  = (const float*)d_in[3];
    const float* bn_var   = (const float*)d_in[4];
    const float* fc1_w    = (const float*)d_in[5];
    const float* fc2_w    = (const float*)d_in[6];
    const float* fc2_b    = (const float*)d_in[7];
    const float* conv_w   = (const float*)d_in[8];
    const int*   inactive = (const int*)d_in[9];
    float* out = (float*)d_out;

    float* wsf   = (float*)d_ws;
    float* xavg  = wsf;                 // 4096
    float* sbn   = wsf + 4096;          // 256
    float* bbn   = wsf + 4352;          // 256
    float* lasso = wsf + 4608;          // 64
    int*   nact  = (int*)(wsf + 4672);  // 64
    int*   aidx  = (int*)(wsf + 4736);  // 16384
    float* aa    = wsf + 4736 + 16384;  // 16384
    float* ab    = aa + 16384;          // 16384

    k1_gap<<<NB * NC, 256, 0, stream>>>(x, bn_gamma, bn_beta, bn_mean, bn_var,
                                        xavg, sbn, bbn);
    k2_gate<<<HEADS * NB, 256, 0, stream>>>(xavg, fc1_w, fc2_w, fc2_b, sbn, bbn,
                                            inactive, lasso, nact, aidx, aa, ab);
    k3_lasso<<<1, 1, 0, stream>>>(lasso, out + (size_t)NB * NC * NHW);
    k4_conv<<<NB * HEADS * 16, 256, 0, stream>>>(x, conv_w, nact, aidx, aa, ab, out);
}
